// LSTM_WP_69999376990622
// MI455X (gfx1250) — matrix operand order repaired and checked by reading, not compiled, not run on hardware
//
#include <hip/hip_runtime.h>
#include <hip/hip_bf16.h>

typedef _Float16 v16h __attribute__((ext_vector_type(16)));
typedef _Float16 v8h  __attribute__((ext_vector_type(8)));
typedef float    v8f  __attribute__((ext_vector_type(8)));
typedef float    v4f  __attribute__((ext_vector_type(4)));

#define HSZ      128
#define BT       16          // batch rows per block (M tile)
#define NTHREADS 256         // 8 wave32 waves
#define GATES    512         // 4*H

// Dynamic LDS layout (bytes):
//   Wf16  : [512][128] f16  = 131072   (W_hh converted once)
//   gsh   : [512][16]  f32  =  32768   (raw gate pre-activations per step)
//   h16   : [16][128]  f16  =   4096   (hidden state, WMMA A operand)
//   h32   : [16][128]  f32  =   8192   (hidden state, full precision)
//   xt    : [16*128]   f32  =   8192   (staged input chunk tile)
#define SMEM_BYTES (131072 + 32768 + 4096 + 8192 + 8192)

__device__ __forceinline__ float sigf(float x) {
    return 1.0f / (1.0f + __expf(-x));
}
__device__ __forceinline__ float tanhfast(float x) {
    return 2.0f / (1.0f + __expf(-2.0f * x)) - 1.0f;
}

__global__ __launch_bounds__(NTHREADS, 1)
void lstm_wmma_kernel(const float* __restrict__ x,
                      const float* __restrict__ W_ih,
                      const float* __restrict__ W_hh,
                      const float* __restrict__ b_ih,
                      const float* __restrict__ b_hh,
                      const float* __restrict__ fc_W,
                      const float* __restrict__ fc_b,
                      float* __restrict__ out)
{
    extern __shared__ char smem[];
    _Float16* Wf16 = (_Float16*)smem;                                  // 512*128 f16
    float*    gsh  = (float*)(smem + 131072);                          // 512*16  f32
    _Float16* h16  = (_Float16*)(smem + 131072 + 32768);               // 16*128  f16
    float*    h32  = (float*)(smem + 131072 + 32768 + 4096);           // 16*128  f32
    float*    xt   = (float*)(smem + 131072 + 32768 + 4096 + 8192);    // 2048    f32

    const int tid  = threadIdx.x;
    const int lane = tid & 31;
    const int wave = tid >> 5;
    const int b0   = blockIdx.x * BT;

    // ---- stage weights (f32 -> f16), input tile, and zero hidden state ----
    for (int i = tid; i < GATES * HSZ; i += NTHREADS)
        Wf16[i] = (_Float16)W_hh[i];
    for (int i = tid; i < BT * HSZ; i += NTHREADS)
        xt[i] = x[(size_t)b0 * HSZ + i];            // contiguous batch chunk
    for (int i = tid; i < BT * HSZ; i += NTHREADS) {
        h16[i] = (_Float16)0.0f;
        h32[i] = 0.0f;
    }
    __syncthreads();

    const int lm = lane & 15;      // N / M selector within tile
    const int lh = lane >> 4;      // lane-half selector (K/M split)

    union Frag { v16h v; v8h h[2]; };

    // ---- preload B-fragments of W_hh^T into registers (loop invariant) ----
    // B[k][n] = W_hh[n][k]; 16-bit B 32x16: lane n = lm, element e -> K = lh*16 + e
    Frag bfrag[4][4];   // [N-tile][K-chunk]
    #pragma unroll
    for (int j = 0; j < 4; ++j) {
        const int n = wave * 64 + j * 16 + lm;
        #pragma unroll
        for (int kc = 0; kc < 4; ++kc) {
            const _Float16* bp = &Wf16[n * HSZ + kc * 32 + lh * 16];
            bfrag[j][kc].h[0] = *(const v8h*)(bp);
            bfrag[j][kc].h[1] = *(const v8h*)(bp + 8);
        }
    }

    // per-lane input-projection constants (input_size == 1)
    float wih[4], bs[4];
    #pragma unroll
    for (int j = 0; j < 4; ++j) {
        const int n = wave * 64 + j * 16 + lm;
        wih[j] = W_ih[n];
        bs[j]  = b_ih[n] + b_hh[n];
    }

    // cell state: thread owns 8 fixed (m,hh) pairs: p = tid + k*256
    float creg[8];
    #pragma unroll
    for (int k = 0; k < 8; ++k) creg[k] = 0.0f;

    for (int t = 0; t < HSZ; ++t) {
        // ================= phase 1: gates = pre + h @ W_hh^T (WMMA) =========
        // 16-bit A 16x32: lane m = lm; elems 0..15 -> K = kc*32 + lh*16 + e
        Frag afrag[4];
        #pragma unroll
        for (int kc = 0; kc < 4; ++kc) {
            const _Float16* ap = &h16[lm * HSZ + kc * 32 + lh * 16];
            afrag[kc].h[0] = *(const v8h*)(ap);
            afrag[kc].h[1] = *(const v8h*)(ap + 8);
        }
        // x values for the 8 accumulator rows this lane touches (m = lh*8 + r)
        float xv[8];
        #pragma unroll
        for (int r = 0; r < 8; ++r)
            xv[r] = xt[(lh * 8 + r) * HSZ + t];

        #pragma unroll
        for (int j = 0; j < 4; ++j) {
            union { v8f v; v4f q[2]; } acc;
            #pragma unroll
            for (int r = 0; r < 8; ++r)
                acc.v[r] = xv[r] * wih[j] + bs[j];   // pre-activation init
            #pragma unroll
            for (int kc = 0; kc < 4; ++kc)
                acc.v = __builtin_amdgcn_wmma_f32_16x16x32_f16(
                    false, afrag[kc].v, false, bfrag[j][kc].v,
                    (short)0, acc.v, false, false);
            // C/D layout: VGPR r -> m = lh*8 + r, n = lm  => contiguous in m
            float* gp = &gsh[(wave * 64 + j * 16 + lm) * BT + lh * 8];
            *(v4f*)(gp)     = acc.q[0];
            *(v4f*)(gp + 4) = acc.q[1];
        }
        __syncthreads();

        // ================= phase 2: activations + state update ==============
        #pragma unroll
        for (int k = 0; k < 8; ++k) {
            const int p  = tid + k * NTHREADS;
            const int hh = p >> 4;
            const int m  = p & 15;
            const float gi = sigf(    gsh[(0 * HSZ + hh) * BT + m]);
            const float gf = sigf(    gsh[(1 * HSZ + hh) * BT + m]);
            const float gg = tanhfast(gsh[(2 * HSZ + hh) * BT + m]);
            const float go = sigf(    gsh[(3 * HSZ + hh) * BT + m]);
            const float c  = gf * creg[k] + gi * gg;
            creg[k] = c;
            const float hv = go * tanhfast(c);
            h32[m * HSZ + hh] = hv;
            h16[m * HSZ + hh] = (_Float16)hv;
        }
        __syncthreads();
    }

    // ================= epilogue: out[b] = h_T . fc_W + fc_b =================
    if (tid < BT) {
        float s = fc_b[0];
        #pragma unroll 8
        for (int k = 0; k < HSZ; ++k)
            s += h32[tid * HSZ + k] * fc_W[k];
        out[b0 + tid] = s;
    }
}

extern "C" void kernel_launch(void* const* d_in, const int* in_sizes, int n_in,
                              void* d_out, int out_size, void* d_ws, size_t ws_size,
                              hipStream_t stream) {
    const float* x    = (const float*)d_in[0];
    const float* W_ih = (const float*)d_in[1];
    const float* W_hh = (const float*)d_in[2];
    const float* b_ih = (const float*)d_in[3];
    const float* b_hh = (const float*)d_in[4];
    const float* fc_W = (const float*)d_in[5];
    const float* fc_b = (const float*)d_in[6];
    float* out = (float*)d_out;

    const int B       = in_sizes[0] / HSZ;   // 4096 chunks
    const int nblocks = B / BT;              // 256 blocks of 16 batches

    hipLaunchKernelGGL(lstm_wmma_kernel, dim3(nblocks), dim3(NTHREADS),
                       SMEM_BYTES, stream,
                       x, W_ih, W_hh, b_ih, b_hh, fc_W, fc_b, out);
    (void)d_ws; (void)ws_size; (void)n_in; (void)out_size;
}
